// AuditableHybridGNN_60103772340852
// MI455X (gfx1250) — matrix operand order repaired and checked by reading, not compiled, not run on hardware
//
#include <hip/hip_runtime.h>

typedef unsigned short u16;
typedef __attribute__((ext_vector_type(16))) __bf16 bf16x16;
typedef __attribute__((ext_vector_type(8)))  float  f32x8;

// ---------- helpers ----------
__device__ __forceinline__ u16 f2bf(float f) {
  unsigned u = __builtin_bit_cast(unsigned, f);
  u += 0x7FFFu + ((u >> 16) & 1u);           // round-to-nearest-even
  return (u16)(u >> 16);
}
__device__ __forceinline__ float bf2f(u16 h) {
  unsigned u = ((unsigned)h) << 16;
  return __builtin_bit_cast(float, u);
}
// monotone float -> uint mapping for atomicMax-based segment max
__device__ __forceinline__ unsigned fmapmax(float f) {
  unsigned u = __builtin_bit_cast(unsigned, f);
  return (u & 0x80000000u) ? ~u : (u | 0x80000000u);
}
__device__ __forceinline__ float funmap(unsigned u) {
  unsigned b = (u & 0x80000000u) ? (u & 0x7FFFFFFFu) : ~u;
  return __builtin_bit_cast(float, b);
}

// ---------- generic WMMA GEMM: C[M,N] = A[M,K] * B^T + bias, B stored as W[n,k] ----------
// block = 4 waves; wave w computes rows [by*256 + w*64, +64) x cols [bx*64, +64)
// requires: M % 256 == 0, N % 64 == 0, K % 32 == 0 (true for all uses here)
__global__ __launch_bounds__(128) void gemm_wmma(
    const u16* __restrict__ A, int lda, long sA,
    const u16* __restrict__ B, int ldb, long sB,
    void* __restrict__ C, int ldc, long sC, int c_bf16,
    const float* __restrict__ bias, int act,
    int M, int N, int K)
{
  const int wave = threadIdx.x >> 5;
  const int lane = threadIdx.x & 31;
  const int half = lane >> 4;
  const int l16  = lane & 15;
  const int m0 = blockIdx.y * 256 + wave * 64;
  const int n0 = blockIdx.x * 64;
  A += (long)blockIdx.z * sA;
  B += (long)blockIdx.z * sB;
  const long coff = (long)blockIdx.z * sC;

  const f32x8 zero8 = {0.f,0.f,0.f,0.f,0.f,0.f,0.f,0.f};
  f32x8 acc[4][4];
#pragma unroll
  for (int mi = 0; mi < 4; ++mi)
#pragma unroll
    for (int ni = 0; ni < 4; ++ni) acc[mi][ni] = zero8;

  for (int k = 0; k < K; k += 32) {
    bf16x16 af[4], bfr[4];
#pragma unroll
    for (int mi = 0; mi < 4; ++mi) {
      // A frag: lane half 0 -> K {k..k+7, k+16..k+23}; half 1 -> +8
      const u16* p = A + (long)(m0 + mi*16 + l16) * lda + k + half*8;
      union { bf16x16 v; uint4 q[2]; } u;
      u.q[0] = *(const uint4*)(p);
      u.q[1] = *(const uint4*)(p + 16);
      af[mi] = u.v;
    }
#pragma unroll
    for (int ni = 0; ni < 4; ++ni) {
      // B frag: lane holds column n = l16; half 0 -> K {k..k+7, k+16..k+23}; half 1 -> +8
      const u16* p = B + (long)(n0 + ni*16 + l16) * ldb + k + half*8;
      union { bf16x16 v; uint4 q[2]; } u;
      u.q[0] = *(const uint4*)(p);
      u.q[1] = *(const uint4*)(p + 16);
      bfr[ni] = u.v;
    }
#pragma unroll
    for (int mi = 0; mi < 4; ++mi)
#pragma unroll
      for (int ni = 0; ni < 4; ++ni)
        acc[mi][ni] = __builtin_amdgcn_wmma_f32_16x16x32_bf16(
            false, af[mi], false, bfr[ni], (short)0, acc[mi][ni], false, false);
  }

#pragma unroll
  for (int mi = 0; mi < 4; ++mi)
#pragma unroll
    for (int ni = 0; ni < 4; ++ni) {
      const int col = n0 + ni*16 + l16;
      const float bv = bias ? bias[col] : 0.f;
#pragma unroll
      for (int r = 0; r < 8; ++r) {
        const int row = m0 + mi*16 + r + 8*half;
        float v = acc[mi][ni][r] + bv;
        if (act == 1) v = v > 0.f ? v : 0.f;
        const long idx = (long)row * ldc + col + coff;
        if (c_bf16) ((u16*)C)[idx] = f2bf(v);
        else        ((float*)C)[idx] = v;
      }
    }
}

// ---------- elementwise / small kernels ----------
__global__ void k_cvt(const float* __restrict__ s, u16* __restrict__ d, long n) {
  long i = (long)blockIdx.x * blockDim.x + threadIdx.x;
  if (i < n) d[i] = f2bf(s[i]);
}
__global__ void k_zero(unsigned* __restrict__ p, long n) {
  long i = (long)blockIdx.x * blockDim.x + threadIdx.x;
  if (i < n) p[i] = 0u;
}
__global__ void k_gelu(const float* __restrict__ s, u16* __restrict__ d, long n) {
  long i = (long)blockIdx.x * blockDim.x + threadIdx.x;
  if (i < n) { float x = s[i]; d[i] = f2bf(0.5f * x * (1.f + erff(x * 0.70710678118f))); }
}
// a[H,64,64] (h,d,e) -> out[h*4096 + e*64 + d] bf16  (W[n,k] form, n=e, k=d)
__global__ void k_relT(const float* __restrict__ a, u16* __restrict__ d) {
  int i = blockIdx.x * blockDim.x + threadIdx.x;
  if (i < 16384) {
    int h = i >> 12, r = (i >> 6) & 63, c = i & 63;
    d[h*4096 + c*64 + r] = f2bf(a[i]);
  }
}
__global__ void k_mix(const float* __restrict__ o, const float* __restrict__ x,
                      const float* __restrict__ skip, float* __restrict__ h, long n) {
  long i = (long)blockIdx.x * blockDim.x + threadIdx.x;
  if (i < n) {
    float s = 1.f / (1.f + __expf(-skip[0]));
    h[i] = s * o[i] + (1.f - s) * x[i];
  }
}
// vT[h*64*4096 + d*4096 + row] = qkv[row*768 + 512 + h*64 + d]
__global__ void k_vtrans(const float* __restrict__ qkv, u16* __restrict__ vt, long n) {
  long i = (long)blockIdx.x * blockDim.x + threadIdx.x;
  if (i < n) {
    long row = i >> 8; int c = (int)(i & 255);
    vt[(long)c * 4096 + row] = f2bf(qkv[row * 768 + 512 + c]);
  }
}

// ---------- HGT edge message passing ----------
// 32 lanes per edge; lane = h*8 + j; j covers 8 dims
__global__ __launch_bounds__(256) void k_edge_logit(
    const float* __restrict__ q, const float* __restrict__ kk, const float* __restrict__ prel,
    const int* __restrict__ src, const int* __restrict__ dst,
    float* __restrict__ logit, unsigned* __restrict__ mx, int E)
{
  int t = blockIdx.x * 256 + threadIdx.x;
  int e = t >> 5;
  if (e >= E) return;
  int l = t & 31, h = l >> 3, j = l & 7;
  int s = src[e], d = dst[e];
  const float* qp = q  + (long)d * 256 + h * 64 + j * 8;
  const float* kp = kk + (long)s * 256 + h * 64 + j * 8;
  float acc = 0.f;
#pragma unroll
  for (int u = 0; u < 8; ++u) acc += qp[u] * kp[u];
  acc += __shfl_down(acc, 4, 8);
  acc += __shfl_down(acc, 2, 8);
  acc += __shfl_down(acc, 1, 8);
  if (j == 0) {
    float lg = acc * prel[h] * 0.125f;          // p_rel / sqrt(64)
    logit[(long)e * 4 + h] = lg;
    atomicMax(&mx[d * 4 + h], fmapmax(lg));
  }
}
__global__ void k_edge_expsum(float* __restrict__ logit, const unsigned* __restrict__ mx,
                              float* __restrict__ sm, const int* __restrict__ dst, long E4) {
  long t = (long)blockIdx.x * blockDim.x + threadIdx.x;
  if (t < E4) {
    long e = t >> 2; int h = (int)(t & 3);
    int d = dst[e];
    float ex = __expf(logit[t] - funmap(mx[d * 4 + h]));
    logit[t] = ex;
    unsafeAtomicAdd(&sm[d * 4 + h], ex);
  }
}
__global__ __launch_bounds__(256) void k_edge_scatter(
    const float* __restrict__ logit, const float* __restrict__ sm, const float* __restrict__ v,
    const int* __restrict__ src, const int* __restrict__ dst, float* __restrict__ agg, int E)
{
  int e = blockIdx.x;
  if (e >= E) return;
  int c = threadIdx.x, h = c >> 6;
  int s = src[e], d = dst[e];
  float a = logit[(long)e * 4 + h] / (sm[d * 4 + h] + 1e-16f);
  unsafeAtomicAdd(&agg[(long)d * 256 + c], v[(long)s * 256 + c] * a);
}

// ---------- row softmax in-place on bf16 matrix (scale 0.125) ----------
__global__ __launch_bounds__(256) void k_softmax_bf(u16* __restrict__ S, int N) {
  __shared__ float red[256];
  long base = (long)blockIdx.x * N;
  int t = threadIdx.x;
  float m = -1e30f;
  for (int c = t; c < N; c += 256) m = fmaxf(m, bf2f(S[base + c]) * 0.125f);
  red[t] = m; __syncthreads();
  for (int o = 128; o; o >>= 1) { if (t < o) red[t] = fmaxf(red[t], red[t + o]); __syncthreads(); }
  m = red[0]; __syncthreads();
  float s = 0.f;
  for (int c = t; c < N; c += 256) s += __expf(bf2f(S[base + c]) * 0.125f - m);
  red[t] = s; __syncthreads();
  for (int o = 128; o; o >>= 1) { if (t < o) red[t] += red[t + o]; __syncthreads(); }
  float inv = 1.f / red[0];
  __syncthreads();
  for (int c = t; c < N; c += 256)
    S[base + c] = f2bf(__expf(bf2f(S[base + c]) * 0.125f - m) * inv);
}

// ---------- LayerNorm of (c1*x1 + c2*x2), dim 256 ----------
__global__ __launch_bounds__(256) void k_ln2(const float* __restrict__ x1, float c1,
                                             const float* __restrict__ x2, float c2,
                                             const float* __restrict__ g, const float* __restrict__ b,
                                             float* __restrict__ out) {
  __shared__ float red[256];
  int t = threadIdx.x;
  long i = (long)blockIdx.x * 256 + t;
  float v = c1 * x1[i] + c2 * x2[i];
  red[t] = v; __syncthreads();
  for (int o = 128; o; o >>= 1) { if (t < o) red[t] += red[t + o]; __syncthreads(); }
  float mean = red[0] * (1.f / 256.f); __syncthreads();
  float dv = v - mean;
  red[t] = dv * dv; __syncthreads();
  for (int o = 128; o; o >>= 1) { if (t < o) red[t] += red[t + o]; __syncthreads(); }
  float var = red[0] * (1.f / 256.f);
  out[i] = dv * rsqrtf(var + 1e-5f) * g[t] + b[t];
}

__global__ __launch_bounds__(256) void k_rel(const float* __restrict__ h, const float* __restrict__ q,
                                             float* __restrict__ rel) {
  __shared__ float red[256];
  int t = threadIdx.x;
  red[t] = h[(long)blockIdx.x * 256 + t] * q[t]; __syncthreads();
  for (int o = 128; o; o >>= 1) { if (t < o) red[t] += red[t + o]; __syncthreads(); }
  if (t == 0) rel[blockIdx.x] = 1.f / (1.f + __expf(-red[0]));
}
__global__ __launch_bounds__(256) void k_ctx(const float* __restrict__ h, const float* __restrict__ rel,
                                             const int* __restrict__ src, const int* __restrict__ dst,
                                             float* __restrict__ ctx, int E) {
  int e = blockIdx.x;
  if (e >= E) return;
  int c = threadIdx.x;
  int s = src[e], d = dst[e];
  unsafeAtomicAdd(&ctx[(long)d * 256 + c], h[(long)s * 256 + c] * rel[s]);
}
__global__ void k_feats(const float* __restrict__ hp, const float* __restrict__ q,
                        u16* __restrict__ f, long n) {
  long i = (long)blockIdx.x * blockDim.x + threadIdx.x;
  if (i < n) {
    long row = i >> 9; int c = (int)(i & 511);
    f[i] = f2bf(c < 256 ? hp[row * 256 + c] : q[c - 256]);
  }
}
__global__ __launch_bounds__(256) void k_score(const float* __restrict__ s1, const float* __restrict__ w2,
                                               const float* __restrict__ b2, float* __restrict__ out) {
  __shared__ float red[256];
  int t = threadIdx.x;
  red[t] = s1[(long)blockIdx.x * 256 + t] * w2[t]; __syncthreads();
  for (int o = 128; o; o >>= 1) { if (t < o) red[t] += red[t + o]; __syncthreads(); }
  if (t == 0) out[blockIdx.x] = red[0] + b2[0];
}

// ---------- host orchestration ----------
extern "C" void kernel_launch(void* const* d_in, const int* in_sizes, int n_in,
                              void* d_out, int out_size, void* d_ws, size_t ws_size,
                              hipStream_t stream) {
  const int DIM = 256;
  const int NEn = in_sizes[0] / DIM;   // 4096 entities
  const int NPn = in_sizes[1] / DIM;   // 4096 passages
  const int E   = in_sizes[39];        // 262144 edges per type
  const int NMX = (NEn > NPn ? NEn : NPn);

  const float* x_ent = (const float*)d_in[0];
  const float* x_psg = (const float*)d_in[1];
  const float* qemb  = (const float*)d_in[2];
  const float *Wk_e=(const float*)d_in[3],  *bk_e=(const float*)d_in[4];
  const float *Wq_e=(const float*)d_in[5],  *bq_e=(const float*)d_in[6];
  const float *Wv_e=(const float*)d_in[7],  *bv_e=(const float*)d_in[8];
  const float *Wo_e=(const float*)d_in[9],  *bo_e=(const float*)d_in[10];
  const float *skip_e=(const float*)d_in[11];
  const float *Wk_p=(const float*)d_in[12], *bk_p=(const float*)d_in[13];
  const float *Wq_p=(const float*)d_in[14], *bq_p=(const float*)d_in[15];
  const float *Wv_p=(const float*)d_in[16], *bv_p=(const float*)d_in[17];
  const float *Wo_p=(const float*)d_in[18], *bo_p=(const float*)d_in[19];
  const float *skip_p=(const float*)d_in[20];
  const float *a_e2p=(const float*)d_in[21], *m_e2p=(const float*)d_in[22], *p_e2p=(const float*)d_in[23];
  const float *a_p2e=(const float*)d_in[24], *m_p2e=(const float*)d_in[25], *p_p2e=(const float*)d_in[26];
  const float *mha_iw=(const float*)d_in[27], *mha_ib=(const float*)d_in[28];
  const float *mha_ow=(const float*)d_in[29], *mha_ob=(const float*)d_in[30];
  const float *ln_eg=(const float*)d_in[31], *ln_eb=(const float*)d_in[32];
  const float *ln_pg=(const float*)d_in[33], *ln_pb=(const float*)d_in[34];
  const float *w1=(const float*)d_in[35], *b1=(const float*)d_in[36];
  const float *w2=(const float*)d_in[37], *b2=(const float*)d_in[38];
  const int *e2p_s=(const int*)d_in[39], *e2p_d=(const int*)d_in[40];
  const int *p2e_s=(const int*)d_in[41], *p2e_d=(const int*)d_in[42];

  char* wsp = (char*)d_ws; size_t off = 0;
  auto alloc = [&](size_t bytes) -> void* {
    off = (off + 255) & ~(size_t)255;
    void* p = wsp + off; off += bytes; return p;
  };
  u16* xe_bf = (u16*)alloc((size_t)NEn*DIM*2);
  u16* xp_bf = (u16*)alloc((size_t)NPn*DIM*2);
  u16 *wkE=(u16*)alloc(131072), *wqE=(u16*)alloc(131072), *wvE=(u16*)alloc(131072), *woE=(u16*)alloc(131072);
  u16 *wkP=(u16*)alloc(131072), *wqP=(u16*)alloc(131072), *wvP=(u16*)alloc(131072), *woP=(u16*)alloc(131072);
  u16* mhaInB  = (u16*)alloc((size_t)768*256*2);
  u16* mhaOutB = (u16*)alloc(131072);
  u16* w1B     = (u16*)alloc((size_t)256*512*2);
  u16 *rtAe2p=(u16*)alloc(32768), *rtMe2p=(u16*)alloc(32768), *rtAp2e=(u16*)alloc(32768), *rtMp2e=(u16*)alloc(32768);
  u16 *keB=(u16*)alloc((size_t)NEn*DIM*2), *veB=(u16*)alloc((size_t)NEn*DIM*2);
  u16 *kpB=(u16*)alloc((size_t)NPn*DIM*2), *vpB=(u16*)alloc((size_t)NPn*DIM*2);
  float *qe=(float*)alloc((size_t)NEn*DIM*4), *qp=(float*)alloc((size_t)NPn*DIM*4);
  float *kprE=(float*)alloc((size_t)NEn*DIM*4), *vprE=(float*)alloc((size_t)NEn*DIM*4);
  float *kprP=(float*)alloc((size_t)NPn*DIM*4), *vprP=(float*)alloc((size_t)NPn*DIM*4);
  float*    logitb=(float*)alloc((size_t)E*4*4);
  unsigned* mxb=(unsigned*)alloc((size_t)NMX*4*4);
  float*    smb=(float*)alloc((size_t)NMX*4*4);
  float *aggP=(float*)alloc((size_t)NPn*DIM*4), *aggE=(float*)alloc((size_t)NEn*DIM*4);
  u16*   gelub=(u16*)alloc((size_t)NMX*DIM*2);
  float* otmp=(float*)alloc((size_t)NMX*DIM*4);
  float *hent=(float*)alloc((size_t)NEn*DIM*4), *hpsg=(float*)alloc((size_t)NPn*DIM*4);
  u16*   hentB=(u16*)alloc((size_t)NEn*DIM*2);
  float* qkv=(float*)alloc((size_t)NEn*768*4);
  u16*   qkvB=(u16*)alloc((size_t)NEn*768*2);
  u16*   vtB =(u16*)alloc((size_t)NEn*DIM*2);
  u16*   Sbf =(u16*)alloc((size_t)NEn*NEn*2);
  float* oatt=(float*)alloc((size_t)NEn*DIM*4);
  u16*   oattB=(u16*)alloc((size_t)NEn*DIM*2);
  float* hglob=(float*)alloc((size_t)NEn*DIM*4);
  float* hent2=(float*)alloc((size_t)NEn*DIM*4);
  float* relb=(float*)alloc((size_t)NEn*4);
  float* ctx=(float*)alloc((size_t)NPn*DIM*4);
  float* hpsg2=(float*)alloc((size_t)NPn*DIM*4);
  u16*   featsB=(u16*)alloc((size_t)NPn*512*2);
  float* s1=(float*)alloc((size_t)NPn*DIM*4);
  (void)ws_size; (void)n_in; (void)out_size;

  auto cvt = [&](const float* s, u16* d, long n) {
    k_cvt<<<dim3((unsigned)((n + 255) / 256)), 256, 0, stream>>>(s, d, n);
  };
  auto zero = [&](void* p, long n) {
    k_zero<<<dim3((unsigned)((n + 255) / 256)), 256, 0, stream>>>((unsigned*)p, n);
  };
  auto gemm = [&](const u16* A, int lda, long sA, const u16* B, int ldb, long sB,
                  void* C, int ldc, long sC, int cbf, const float* bias, int act,
                  int M, int N, int K, int batch) {
    dim3 g((unsigned)(N / 64), (unsigned)(M / 256), (unsigned)batch);
    gemm_wmma<<<g, 128, 0, stream>>>(A, lda, sA, B, ldb, sB, C, ldc, sC, cbf, bias, act, M, N, K);
  };

  // ---- conversions ----
  cvt(x_ent, xe_bf, (long)NEn*DIM);  cvt(x_psg, xp_bf, (long)NPn*DIM);
  cvt(Wk_e, wkE, 65536); cvt(Wq_e, wqE, 65536); cvt(Wv_e, wvE, 65536); cvt(Wo_e, woE, 65536);
  cvt(Wk_p, wkP, 65536); cvt(Wq_p, wqP, 65536); cvt(Wv_p, wvP, 65536); cvt(Wo_p, woP, 65536);
  cvt(mha_iw, mhaInB, 768*256); cvt(mha_ow, mhaOutB, 65536); cvt(w1, w1B, 256*512);
  k_relT<<<64, 256, 0, stream>>>(a_e2p, rtAe2p);
  k_relT<<<64, 256, 0, stream>>>(m_e2p, rtMe2p);
  k_relT<<<64, 256, 0, stream>>>(a_p2e, rtAp2e);
  k_relT<<<64, 256, 0, stream>>>(m_p2e, rtMp2e);

  // ---- KQV projections (WMMA) ----
  gemm(xe_bf, 256, 0, wkE, 256, 0, keB, 256, 0, 1, bk_e, 0, NEn, 256, 256, 1);
  gemm(xe_bf, 256, 0, wqE, 256, 0, qe,  256, 0, 0, bq_e, 0, NEn, 256, 256, 1);
  gemm(xe_bf, 256, 0, wvE, 256, 0, veB, 256, 0, 1, bv_e, 0, NEn, 256, 256, 1);
  gemm(xp_bf, 256, 0, wkP, 256, 0, kpB, 256, 0, 1, bk_p, 0, NPn, 256, 256, 1);
  gemm(xp_bf, 256, 0, wqP, 256, 0, qp,  256, 0, 0, bq_p, 0, NPn, 256, 256, 1);
  gemm(xp_bf, 256, 0, wvP, 256, 0, vpB, 256, 0, 1, bv_p, 0, NPn, 256, 256, 1);

  // ---- relation transforms, batched over 4 heads ----
  gemm(keB, 256, 64, rtAe2p, 64, 4096, kprE, 256, 64, 0, nullptr, 0, NEn, 64, 64, 4);
  gemm(veB, 256, 64, rtMe2p, 64, 4096, vprE, 256, 64, 0, nullptr, 0, NEn, 64, 64, 4);
  gemm(kpB, 256, 64, rtAp2e, 64, 4096, kprP, 256, 64, 0, nullptr, 0, NPn, 64, 64, 4);
  gemm(vpB, 256, 64, rtMp2e, 64, 4096, vprP, 256, 64, 0, nullptr, 0, NPn, 64, 64, 4);

  // ---- edge type e2p: entity -> passage ----
  zero(mxb, (long)NPn*4); zero(smb, (long)NPn*4);
  k_edge_logit<<<dim3((unsigned)(E/8)), 256, 0, stream>>>(qp, kprE, p_e2p, e2p_s, e2p_d, logitb, mxb, E);
  k_edge_expsum<<<dim3((unsigned)((E*4+255)/256)), 256, 0, stream>>>(logitb, mxb, smb, e2p_d, (long)E*4);
  zero(aggP, (long)NPn*DIM);
  k_edge_scatter<<<dim3((unsigned)E), 256, 0, stream>>>(logitb, smb, vprE, e2p_s, e2p_d, aggP, E);
  // ---- edge type p2e: passage -> entity ----
  zero(mxb, (long)NEn*4); zero(smb, (long)NEn*4);
  k_edge_logit<<<dim3((unsigned)(E/8)), 256, 0, stream>>>(qe, kprP, p_p2e, p2e_s, p2e_d, logitb, mxb, E);
  k_edge_expsum<<<dim3((unsigned)((E*4+255)/256)), 256, 0, stream>>>(logitb, mxb, smb, p2e_d, (long)E*4);
  zero(aggE, (long)NEn*DIM);
  k_edge_scatter<<<dim3((unsigned)E), 256, 0, stream>>>(logitb, smb, vprP, p2e_s, p2e_d, aggE, E);

  // ---- HGT output: h = sig(skip)*(gelu(agg)@Wout^T+b) + (1-sig)*x ----
  k_gelu<<<dim3((unsigned)((NEn*DIM+255)/256)), 256, 0, stream>>>(aggE, gelub, (long)NEn*DIM);
  gemm(gelub, 256, 0, woE, 256, 0, otmp, 256, 0, 0, bo_e, 0, NEn, 256, 256, 1);
  k_mix<<<dim3((unsigned)((NEn*DIM+255)/256)), 256, 0, stream>>>(otmp, x_ent, skip_e, hent, (long)NEn*DIM);
  k_gelu<<<dim3((unsigned)((NPn*DIM+255)/256)), 256, 0, stream>>>(aggP, gelub, (long)NPn*DIM);
  gemm(gelub, 256, 0, woP, 256, 0, otmp, 256, 0, 0, bo_p, 0, NPn, 256, 256, 1);
  k_mix<<<dim3((unsigned)((NPn*DIM+255)/256)), 256, 0, stream>>>(otmp, x_psg, skip_p, hpsg, (long)NPn*DIM);

  // ---- global MHA over entities ----
  cvt(hent, hentB, (long)NEn*DIM);
  gemm(hentB, 256, 0, mhaInB, 256, 0, qkv, 768, 0, 0, mha_ib, 0, NEn, 768, 256, 1);
  cvt(qkv, qkvB, (long)NEn*768);
  k_vtrans<<<dim3((unsigned)((NEn*DIM+255)/256)), 256, 0, stream>>>(qkv, vtB, (long)NEn*DIM);
  for (int h = 0; h < 4; ++h) {
    // S = Q_h K_h^T (bf16 out)
    gemm(qkvB + h*64, 768, 0, qkvB + 256 + h*64, 768, 0, Sbf, NEn, 0, 1, nullptr, 0, NEn, NEn, 64, 1);
    k_softmax_bf<<<dim3((unsigned)NEn), 256, 0, stream>>>(Sbf, NEn);
    // O_h = P V_h (via pre-transposed V)
    gemm(Sbf, NEn, 0, vtB + (long)h*64*NEn, NEn, 0, oatt + h*64, 256, 0, 0, nullptr, 0, NEn, 64, NEn, 1);
  }
  cvt(oatt, oattB, (long)NEn*DIM);
  gemm(oattB, 256, 0, mhaOutB, 256, 0, hglob, 256, 0, 0, mha_ob, 0, NEn, 256, 256, 1);
  k_ln2<<<dim3((unsigned)NEn), 256, 0, stream>>>(hent, 0.9f, hglob, 0.1f, ln_eg, ln_eb, hent2);

  // ---- gather-scale-scatter entity -> passage ----
  k_rel<<<dim3((unsigned)NEn), 256, 0, stream>>>(hent2, qemb, relb);
  zero(ctx, (long)NPn*DIM);
  k_ctx<<<dim3((unsigned)E), 256, 0, stream>>>(hent2, relb, e2p_s, e2p_d, ctx, E);
  k_ln2<<<dim3((unsigned)NPn), 256, 0, stream>>>(hpsg, 1.f, ctx, 1.f, ln_pg, ln_pb, hpsg2);

  // ---- scoring head ----
  k_feats<<<dim3((unsigned)((NPn*512+255)/256)), 256, 0, stream>>>(hpsg2, qemb, featsB, (long)NPn*512);
  gemm(featsB, 512, 0, w1B, 512, 0, s1, 256, 0, 0, b1, 1 /*relu*/, NPn, 256, 512, 1);
  k_score<<<dim3((unsigned)NPn), 256, 0, stream>>>(s1, w2, b2, (float*)d_out);
}
